// AttentionLayer_6665789243987
// MI455X (gfx1250) — matrix operand order repaired and checked by reading, not compiled, not run on hardware
//
#include <hip/hip_runtime.h>

// Problem constants (match reference)
#define B_   16
#define E_   512
#define DIN_ 1024
#define H_   8
#define D_   128
#define T_   3
#define NEG_ (-1e30f)

typedef __attribute__((ext_vector_type(16))) __bf16 v16bf;
typedef __attribute__((ext_vector_type(8)))  __bf16 v8bf;
typedef __attribute__((ext_vector_type(8)))  float  v8f;

__device__ __forceinline__ unsigned short f2bfu(float f) {
  unsigned u = __builtin_bit_cast(unsigned, f);
  return (unsigned short)((u + 0x7fffu + ((u >> 16) & 1u)) >> 16); // RNE
}
__device__ __forceinline__ __bf16 f2bf(float f) {
  unsigned short s = f2bfu(f);
  return __builtin_bit_cast(__bf16, s);
}
// Pack two f32 -> packed 2xbf16 (single v_cvt_pk_bf16_f32 where available)
__device__ __forceinline__ unsigned pk2bf(float a, float b) {
#if __has_builtin(__builtin_amdgcn_cvt_pk_bf16_f32)
  auto v = __builtin_amdgcn_cvt_pk_bf16_f32(a, b);
  return __builtin_bit_cast(unsigned, v);
#else
  return (unsigned)f2bfu(a) | ((unsigned)f2bfu(b) << 16);
#endif
}
__device__ __forceinline__ v8f zero8() {
  v8f z = {0.f,0.f,0.f,0.f,0.f,0.f,0.f,0.f};
  return z;
}

// CDNA5 async global->LDS copy (GV mode, 16B per lane), tracked by ASYNCcnt.
__device__ __forceinline__ void async_copy_b128(unsigned lds_off, const void* gptr) {
  asm volatile("global_load_async_to_lds_b128 %0, %1, off"
               :: "v"(lds_off), "v"((unsigned long long)(uintptr_t)gptr)
               : "memory");
}
__device__ __forceinline__ void wait_async0() {
  asm volatile("s_wait_asynccnt 0x0" ::: "memory");
}
__device__ __forceinline__ void wait_async2() {  // allow 2 transfers in flight
  asm volatile("s_wait_asynccnt 0x2" ::: "memory");
}

// ---------------------------------------------------------------------------
// Kernel 0: x (f32) -> xbf (bf16), enables pure-copy async staging in the GEMM
// ---------------------------------------------------------------------------
__global__ __launch_bounds__(256) void k_cvt(const float* __restrict__ x,
                                             unsigned short* __restrict__ xbf) {
  const float4* src = (const float4*)x;
  uint2* dst = (uint2*)xbf;
  const int nq = (B_ * E_ * DIN_) / 4;
  for (int q = blockIdx.x * 256 + threadIdx.x; q < nq; q += 2048 * 256) {
    float4 f = src[q];
    uint2 p;
    p.x = pk2bf(f.x, f.y);
    p.y = pk2bf(f.z, f.w);
    dst[q] = p;
  }
}

// ---------------------------------------------------------------------------
// Kernel A: fold attention vectors into W:  v1[ht][i] = sum_d W[ht][i][d]*a1[ht][d]
// ---------------------------------------------------------------------------
__global__ __launch_bounds__(128) void k_fold(const float* __restrict__ W,
                                              const float* __restrict__ a1,
                                              const float* __restrict__ a2,
                                              float* __restrict__ v1,
                                              float* __restrict__ v2) {
  __shared__ float a1s[D_], a2s[D_];
  const int ht  = blockIdx.x;
  const int tid = threadIdx.x;
  a1s[tid] = a1[ht * D_ + tid];
  a2s[tid] = a2[ht * D_ + tid];
  __syncthreads();
  for (int i = tid; i < DIN_; i += 128) {
    const float* wr = W + (size_t)(ht * DIN_ + i) * D_;
    float s1 = 0.f, s2 = 0.f;
    for (int d = 0; d < D_; ++d) {
      float w = wr[d];
      s1 += w * a1s[d];
      s2 += w * a2s[d];
    }
    v1[ht * DIN_ + i] = s1;
    v2[ht * DIN_ + i] = s2;
  }
}

// ---------------------------------------------------------------------------
// Kernel B: left[ht][row] = x[row,:]·v1[ht], right[ht][row] = x[row,:]·v2[ht]
// x row staged in LDS via async b128 copies; wave32 shuffle reduction.
// ---------------------------------------------------------------------------
__global__ __launch_bounds__(256) void k_leftright(const float* __restrict__ x,
                                                   const float* __restrict__ v1,
                                                   const float* __restrict__ v2,
                                                   float* __restrict__ left,
                                                   float* __restrict__ right) {
  __shared__ float xs[DIN_];
  const int row = blockIdx.x;        // b*E + e
  const int tid = threadIdx.x;
  {
    unsigned lbase = (unsigned)(uintptr_t)(&xs[0]);
    const char* g = (const char*)(x + (size_t)row * DIN_);
    async_copy_b128(lbase + tid * 16, g + tid * 16);   // 256 * 16B = 4KB
    wait_async0();
  }
  __syncthreads();
  const int wave = tid >> 5, lane = tid & 31;
  for (int o = wave; o < 2 * H_ * T_; o += 8) {   // 48 outputs over 8 waves
    const int ht = o >> 1;
    const float* v = ((o & 1) ? v2 : v1) + ht * DIN_;
    float p = 0.f;
    for (int i = lane; i < DIN_; i += 32) p += xs[i] * v[i];
    for (int off = 16; off > 0; off >>= 1) p += __shfl_xor(p, off, 32);
    if (lane == 0) {
      float* dst = (o & 1) ? right : left;
      dst[ht * (B_ * E_) + row] = p;
    }
  }
}

// ---------------------------------------------------------------------------
// Kernel C: h_last GEMM (bf16 WMMA): hl_t[h][b][d][e] = sum_i x[b,e,i]*W[h,T-1,i,d]
// Double-buffered: async b128 copy of A tile k+1 stays in flight while tile k
// is computed (s_wait_asynccnt 2). B tile staged transposed ([d][k]) so B
// fragments are two contiguous 16B v8bf loads each. Output bf16, transposed [d][e].
// ---------------------------------------------------------------------------
__global__ __launch_bounds__(256) void k_hlast(const unsigned short* __restrict__ xbf,
                                               const float* __restrict__ W,
                                               const float* __restrict__ mask,
                                               unsigned short* __restrict__ hl_t) {
  __shared__ __bf16 As[2][128 * 32];    // [buf][row][k]
  __shared__ __bf16 Bst[2][128 * 32];   // [buf][d][k] (transposed)
  const int mtile = blockIdx.x, h = blockIdx.y;
  const int tid = threadIdx.x, wave = tid >> 5, lane = tid & 31;
  const int row0 = mtile * 128;
  const float* Wp = W + (size_t)((h * T_ + (T_ - 1)) * DIN_) * D_;  // last edge type
  const int wm = wave >> 1, wn = wave & 1;
  const unsigned asb = (unsigned)(uintptr_t)(&As[0][0]);

  auto issue_A = [&](int k0n, int buf) {
#pragma unroll
    for (int it = 0; it < 2; ++it) {
      int idx = tid + 256 * it;
      int r = idx >> 2, kq = idx & 3;
      const char* g = (const char*)xbf +
          (((size_t)(row0 + r) * DIN_ + k0n + kq * 8) << 1);
      async_copy_b128(asb + (unsigned)(buf * 8192 + r * 64 + kq * 16), g);
    }
  };

  v8f acc[2][4];
  for (int i = 0; i < 2; ++i)
    for (int j = 0; j < 4; ++j) acc[i][j] = zero8();

  issue_A(0, 0);                       // prologue: tile 0 in flight

  const int NK = DIN_ / 32;            // 32 K-tiles
  for (int kt = 0; kt < NK; ++kt) {
    const int k0 = kt * 32;
    const int buf = kt & 1;
    if (kt + 1 < NK) issue_A(k0 + 32, buf ^ 1);   // next A tile in flight

    // --- B tile: read 4 k-values per thread (coalesced over d), convert,
    //     store transposed as one b64 per thread-iter
    for (int s = 0; s < 4; ++s) {
      int q = tid + 256 * s;           // 1024 quads
      int d = q & 127, kq = q >> 7;    // kq in 0..7
      const float* wp = Wp + (size_t)(k0 + kq * 4) * D_ + d;
      float f0 = wp[0 * D_], f1 = wp[1 * D_], f2 = wp[2 * D_], f3 = wp[3 * D_];
      uint2 p;
      p.x = pk2bf(f0, f1);
      p.y = pk2bf(f2, f3);
      *(uint2*)&Bst[buf][d * 32 + kq * 4] = p;
    }
    if (kt + 1 < NK) {
      __builtin_prefetch(Wp + (size_t)(k0 + 32 + (tid & 31)) * D_, 0, 0);
      wait_async2();                   // tile k done; tile k+1 still in flight
    } else {
      wait_async0();
    }
    __syncthreads();

    // --- fragments ---
    const int l15 = lane & 15;
    const int koff = (lane < 16) ? 0 : 8;   // A per-lane K offset
    const int kb   = (lane < 16) ? 0 : 8;   // B per-lane K offset
    v16bf a[2];
    for (int mf = 0; mf < 2; ++mf) {
      int r = wm * 32 + mf * 16 + l15;
      v8bf lo = *(const v8bf*)&As[buf][r * 32 + koff];
      v8bf hi = *(const v8bf*)&As[buf][r * 32 + 16 + koff];
      a[mf] = __builtin_shufflevector(lo, hi, 0,1,2,3,4,5,6,7,8,9,10,11,12,13,14,15);
    }
    for (int nf = 0; nf < 4; ++nf) {
      int c = wn * 64 + nf * 16 + l15;
      v8bf blo = *(const v8bf*)&Bst[buf][c * 32 + kb];
      v8bf bhi = *(const v8bf*)&Bst[buf][c * 32 + 16 + kb];
      v16bf bfr = __builtin_shufflevector(blo, bhi, 0,1,2,3,4,5,6,7,8,9,10,11,12,13,14,15);
      for (int mf = 0; mf < 2; ++mf)
        acc[mf][nf] = __builtin_amdgcn_wmma_f32_16x16x32_bf16(
            false, a[mf], false, bfr, (short)0, acc[mf][nf], false, false);
    }
    __syncthreads();   // all waves done reading buf before it is re-filled
  }

  // Store transposed bf16: lane owns 8 consecutive e-rows -> one b128 store
  const int l15s = lane & 15;
  const int mo = (lane < 16) ? 0 : 8;
  for (int mf = 0; mf < 2; ++mf) {
    int m0 = wm * 32 + mf * 16 + mo;          // first of 8 consecutive rows
    int grow0 = row0 + m0;
    int bb = grow0 >> 9, e0 = grow0 & 511;    // 8-row group never crosses b
    float4 ma = *(const float4*)&mask[grow0];
    float4 mb = *(const float4*)&mask[grow0 + 4];
    for (int nf = 0; nf < 4; ++nf) {
      int n = wn * 64 + nf * 16 + l15s;
      const v8f& v = acc[mf][nf];
      uint4 pk;
      pk.x = pk2bf(v[0] * ma.x, v[1] * ma.y);
      pk.y = pk2bf(v[2] * ma.z, v[3] * ma.w);
      pk.z = pk2bf(v[4] * mb.x, v[5] * mb.y);
      pk.w = pk2bf(v[6] * mb.z, v[7] * mb.w);
      *(uint4*)&hl_t[(((size_t)(h * B_ + bb) * D_ + n) * E_) + e0] = pk;
    }
  }
}

// ---------------------------------------------------------------------------
// Kernel D: per (h, b, 32-row tile): scores -> f32 softmax -> WMMA aggregation.
// B fragments read directly from global hl_t (L2-resident), no K-loop barriers.
// ---------------------------------------------------------------------------
__global__ __launch_bounds__(256) void k_attn(const int* __restrict__ adj,
                                              const float* __restrict__ left,
                                              const float* __restrict__ right,
                                              const unsigned short* __restrict__ hl_t,
                                              float* __restrict__ out) {
  extern __shared__ char dsm[];
  float*  sS   = (float*)dsm;              // 32*512 scores (f32)
  float*  sR   = sS + 32 * E_;             // 3*512 right
  float*  sL   = sR + T_ * E_;             // 3*32 left
  float*  sM   = sL + T_ * 32;             // 32*8 partial max
  float*  sSum = sM + 32 * 8;              // 32*8 partial sum
  __bf16* sP   = (__bf16*)(sSum + 32 * 8); // 32*512 probs (bf16)

  const int etile = blockIdx.x, b = blockIdx.y, h = blockIdx.z;
  const int tid = threadIdx.x, wave = tid >> 5, lane = tid & 31;
  const int i0 = etile * 32;

  if (tid < T_ * 32) {
    int t = tid >> 5, r = tid & 31;
    sL[t * 32 + r] = left[(h * T_ + t) * (B_ * E_) + b * E_ + i0 + r];
  }
  for (int idx = tid; idx < T_ * E_; idx += 256) {
    int t = idx >> 9, j = idx & 511;
    sR[idx] = right[(h * T_ + t) * (B_ * E_) + b * E_ + j];
  }
  __syncthreads();

  // Scores: adj picks exactly one edge type
  const int* adjp = adj + (size_t)b * E_ * E_ + (size_t)i0 * E_;
  for (int idx = tid; idx < 32 * E_; idx += 256) {
    int r = idx >> 9, j = idx & 511;
    int a = adjp[r * E_ + j];
    float s;
    if (a > 0) {
      float v = sL[(a - 1) * 32 + r] + sR[(a - 1) * E_ + j];
      s = (v > 0.f) ? v : 0.2f * v;
    } else {
      s = NEG_;
    }
    sS[r * E_ + j] = s;
  }
  __syncthreads();

  // Row softmax in f32: 8 threads per row, 64 elems per thread
  {
    int r = tid >> 3, seg = tid & 7;
    float* rowp = sS + r * E_;
    int j0 = seg * 64;
    float m = -3.4e38f;
    for (int jj = 0; jj < 64; ++jj) m = fmaxf(m, rowp[j0 + jj]);
    sM[r * 8 + seg] = m;
    __syncthreads();
    float rm = sM[r * 8];
    for (int q = 1; q < 8; ++q) rm = fmaxf(rm, sM[r * 8 + q]);
    float lsum = 0.f;
    for (int jj = 0; jj < 64; ++jj) {
      float ev = __expf(rowp[j0 + jj] - rm);
      rowp[j0 + jj] = ev;
      lsum += ev;
    }
    sSum[r * 8 + seg] = lsum;
    __syncthreads();
    float rs = 0.f;
    for (int q = 0; q < 8; ++q) rs += sSum[r * 8 + q];
    float inv = 1.f / rs;
    for (int jj = 0; jj < 64; jj += 2) {   // pairwise packed bf16 stores
      unsigned pk = pk2bf(rowp[j0 + jj] * inv, rowp[j0 + jj + 1] * inv);
      *(unsigned*)&sP[r * E_ + j0 + jj] = pk;
    }
  }
  __syncthreads();

  // Aggregation: P[32x512](bf16, LDS) x hl_t[h,b][128x512](bf16, global/L2)
  const int mfrag = wave & 1;          // 16-row half
  const int nbase = (wave >> 1) * 32;  // 32-col slab (2 frags)
  v8f acc0 = zero8(), acc1 = zero8();
  const int l15 = lane & 15;
  const int koff = (lane < 16) ? 0 : 8;
  const int kb   = (lane < 16) ? 0 : 8;
  const __bf16* hb = (const __bf16*)hl_t + ((size_t)(h * B_ + b) * D_) * E_;
  const __bf16* bp0 = hb + (size_t)(nbase + 0 * 16 + l15) * E_ + kb;
  const __bf16* bp1 = hb + (size_t)(nbase + 1 * 16 + l15) * E_ + kb;
  const __bf16* ap  = sP + (mfrag * 16 + l15) * E_;

  for (int kk = 0; kk < E_ / 32; ++kk) {
    v8bf lo = *(const v8bf*)&ap[kk * 32 + koff];
    v8bf hi = *(const v8bf*)&ap[kk * 32 + 16 + koff];
    v16bf a = __builtin_shufflevector(lo, hi, 0,1,2,3,4,5,6,7,8,9,10,11,12,13,14,15);
    v16bf b0 = __builtin_shufflevector(*(const v8bf*)&bp0[kk * 32], *(const v8bf*)&bp0[kk * 32 + 16], 0,1,2,3,4,5,6,7,8,9,10,11,12,13,14,15);
    v16bf b1 = __builtin_shufflevector(*(const v8bf*)&bp1[kk * 32], *(const v8bf*)&bp1[kk * 32 + 16], 0,1,2,3,4,5,6,7,8,9,10,11,12,13,14,15);
    acc0 = __builtin_amdgcn_wmma_f32_16x16x32_bf16(false, a, false, b0,
                                                   (short)0, acc0, false, false);
    acc1 = __builtin_amdgcn_wmma_f32_16x16x32_bf16(false, a, false, b1,
                                                   (short)0, acc1, false, false);
  }

  // Store with ReLU, concat heads: out[b, e, h*128 + n]; NT (write-once data)
  const int mo = (lane < 16) ? 0 : 8;
  for (int nf = 0; nf < 2; ++nf) {
    v8f accv = nf ? acc1 : acc0;
    for (int e = 0; e < 8; ++e) {
      int m = mfrag * 16 + e + mo;
      int n = nbase + nf * 16 + l15;
      int eg = i0 + m;
      __builtin_nontemporal_store(
          fmaxf(accv[e], 0.f),
          &out[((size_t)(b * E_ + eg)) * (H_ * D_) + h * D_ + n]);
    }
  }
}

// ---------------------------------------------------------------------------
extern "C" void kernel_launch(void* const* d_in, const int* in_sizes, int n_in,
                              void* d_out, int out_size, void* d_ws, size_t ws_size,
                              hipStream_t stream) {
  (void)in_sizes; (void)n_in; (void)out_size; (void)ws_size;
  const float* x    = (const float*)d_in[0];
  const int*   adj  = (const int*)d_in[1];
  const float* mask = (const float*)d_in[2];
  const float* W    = (const float*)d_in[3];
  const float* a1   = (const float*)d_in[4];
  const float* a2   = (const float*)d_in[5];
  float* out = (float*)d_out;

  // Workspace: v1,v2 + left,right (f32) + hl_t (bf16, 16MB) + xbf (bf16, 16MB)
  float* v1 = (float*)d_ws;
  float* v2 = v1 + H_ * T_ * DIN_;
  float* left = v2 + H_ * T_ * DIN_;
  float* right = left + H_ * T_ * B_ * E_;
  unsigned short* hl  = (unsigned short*)(right + H_ * T_ * B_ * E_);
  unsigned short* xbf = hl + (size_t)H_ * B_ * D_ * E_;

  hipLaunchKernelGGL(k_cvt, dim3(2048), dim3(256), 0, stream, x, xbf);
  hipLaunchKernelGGL(k_fold, dim3(H_ * T_), dim3(128), 0, stream, W, a1, a2, v1, v2);
  hipLaunchKernelGGL(k_leftright, dim3(B_ * E_), dim3(256), 0, stream, x, v1, v2, left, right);
  hipLaunchKernelGGL(k_hlast, dim3((B_ * E_) / 128, H_), dim3(256), 0, stream, xbf, W, mask, hl);

  size_t smem = (size_t)(32 * E_ + T_ * E_ + T_ * 32 + 32 * 8 + 32 * 8) * sizeof(float)
              + (size_t)(32 * E_) * sizeof(unsigned short); // ~104.5 KB
  hipLaunchKernelGGL(k_attn, dim3(E_ / 32, B_, H_), dim3(256), smem, stream,
                     adj, left, right, hl, out);
}
